// Embedder_57303453663628
// MI455X (gfx1250) — matrix operand order repaired and checked by reading, not compiled, not run on hardware
//
#include <hip/hip_runtime.h>
#include <hip/hip_bf16.h>

#define BDIM 4096
#define GDIM 20000
#define HDIM 512
#define LN_EPS 1e-5f

typedef __attribute__((ext_vector_type(16))) __bf16        v16bf;
typedef __attribute__((ext_vector_type(8)))  float         v8f;
typedef __attribute__((ext_vector_type(8)))  unsigned int  v8u;
typedef __attribute__((ext_vector_type(4)))  unsigned int  v4u;

// f32 pair -> packed bf16 (round half-up): one v_add each + one v_perm_b32.
// v_perm_b32 byte-vector is {S0=hi : bytes 7..4, S1=lo : bytes 3..0};
// sel 0x07060302 -> dst = {hi[31:16], lo[31:16]}.
__device__ __forceinline__ unsigned int pack_bf16(float lo, float hi) {
    unsigned int ul = __float_as_uint(lo) + 0x8000u;
    unsigned int uh = __float_as_uint(hi) + 0x8000u;
    return __builtin_amdgcn_perm(uh, ul, 0x07060302u);
}

// -------------------------------------------------------------------------
// GEMM: out[4096,512] = x[4096,20000] @ emb[20000,512]
// bf16 WMMA, f32 accumulate. WG tile 128x64, wave tile 32x32.
// BK=160 (= 5 WMMA K-substeps) -> 20 WMMAs per barrier pair, 125 iterations.
// -------------------------------------------------------------------------
#define BM 128
#define BN 64
#define BK 160
#define KSUB (BK / 32)          // 5 WMMA sub-steps per staged tile
#define NPAIR (BK / 2)          // 80 packed bf16 pairs per LDS row
#define APITCH (NPAIR + 4)      // 84 dwords: +4 pad, keeps 16B alignment

__global__ __launch_bounds__(256)
void gemm_xemb_bf16(const float* __restrict__ x,
                    const float* __restrict__ emb,
                    float* __restrict__ out) {
    __shared__ __align__(16) unsigned int Asl[BM * APITCH]; // [m][kpair]
    __shared__ __align__(16) unsigned int Bsl[BN * APITCH]; // [n][kpair]

    const int tid  = threadIdx.x;
    const int lane = tid & 31;
    const int wave = tid >> 5;
    const int lr   = lane & 15;   // row-within-fragment
    const int h    = lane >> 4;   // lane half (K interleave select)

    const int m0 = (wave & 3) * 32;   // wave row block
    const int n0 = (wave >> 2) * 32;  // wave col block

    const int mt = blockIdx.x >> 3;   // 0..31
    const int nt = blockIdx.x & 7;    // 0..7
    const int row0 = mt * BM;
    const int col0 = nt * BN;

    v8f acc[2][2];
#pragma unroll
    for (int i = 0; i < 2; ++i)
#pragma unroll
        for (int j = 0; j < 2; ++j)
            acc[i][j] = (v8f)(0.0f);

    // B-tile staging map: thread -> (column n, 20 consecutive k-pairs)
    const int bn  = tid & 63;
    const int bkp = (tid >> 6) * (NPAIR / 4);

#pragma unroll 1
    for (int kt = 0; kt < GDIM / BK; ++kt) {
        const int k0 = kt * BK;

        // ---- stage A tile: 128 x 160 fp32 -> bf16 pairs in LDS ----
        // flat float4 index over [128 rows][40 float4/row], coalesced.
#pragma unroll
        for (int q = 0; q < (BM * BK / 4) / 256; ++q) {  // 20
            const int idx = tid + q * 256;
            const int r   = idx / (BK / 4);
            const int c4  = idx - r * (BK / 4);
            const float4 av = *reinterpret_cast<const float4*>(
                x + (size_t)(row0 + r) * GDIM + k0 + c4 * 4);
            Asl[r * APITCH + c4 * 2 + 0] = pack_bf16(av.x, av.y);
            Asl[r * APITCH + c4 * 2 + 1] = pack_bf16(av.z, av.w);
        }

        // ---- stage B tile: 160 x 64 fp32 -> bf16 pairs in LDS (transposed) ----
#pragma unroll
        for (int p = 0; p < NPAIR / 4; ++p) {            // 20
            const int kp = bkp + p;
            const int k  = kp * 2;
            const float f0 = emb[(size_t)(k0 + k)     * HDIM + col0 + bn];
            const float f1 = emb[(size_t)(k0 + k + 1) * HDIM + col0 + bn];
            Bsl[bn * APITCH + kp] = pack_bf16(f0, f1);
        }

        __syncthreads();

        // ---- 5 K-substeps x 4 WMMAs, fragments via aligned ds_load_b128 ----
#pragma unroll
        for (int s = 0; s < KSUB; ++s) {
            v8u afrag[2], bfrag[2];
#pragma unroll
            for (int i = 0; i < 2; ++i) {
                const unsigned int* arow =
                    &Asl[(m0 + 16 * i + lr) * APITCH + 16 * s + 4 * h];
                const v4u lo = *reinterpret_cast<const v4u*>(arow);
                const v4u hi = *reinterpret_cast<const v4u*>(arow + 8);
                afrag[i] = (v8u){lo.x, lo.y, lo.z, lo.w, hi.x, hi.y, hi.z, hi.w};
            }
#pragma unroll
            for (int j = 0; j < 2; ++j) {
                const unsigned int* brow =
                    &Bsl[(n0 + 16 * j + lr) * APITCH + 16 * s + 4 * h];
                const v4u lo = *reinterpret_cast<const v4u*>(brow);
                const v4u hi = *reinterpret_cast<const v4u*>(brow + 8);
                bfrag[j] = (v8u){lo.x, lo.y, lo.z, lo.w, hi.x, hi.y, hi.z, hi.w};
            }
#pragma unroll
            for (int i = 0; i < 2; ++i)
#pragma unroll
                for (int j = 0; j < 2; ++j)
                    acc[i][j] = __builtin_amdgcn_wmma_f32_16x16x32_bf16(
                        false, __builtin_bit_cast(v16bf, afrag[i]),
                        false, __builtin_bit_cast(v16bf, bfrag[j]),
                        (short)0, acc[i][j], false, false);
        }

        __syncthreads();
    }

    // ---- store raw accumulators (pre-GELU/LN) ----
    // C/D layout: VGPR r, lanes 0-15 -> M=r, lanes 16-31 -> M=8+r, N=lane%16
#pragma unroll
    for (int i = 0; i < 2; ++i)
#pragma unroll
        for (int j = 0; j < 2; ++j) {
            float* base = out + (size_t)(row0 + m0 + 16 * i + 8 * h) * HDIM
                              + (col0 + n0 + 16 * j + lr);
#pragma unroll
            for (int r = 0; r < 8; ++r)
                base[(size_t)r * HDIM] = acc[i][j][r];
        }
}

// -------------------------------------------------------------------------
// Epilogue: exact GELU + LayerNorm, in-place on out. One WG per row.
// -------------------------------------------------------------------------
__device__ __forceinline__ float gelu_exact(float v) {
    return 0.5f * v * (1.0f + erff(v * 0.70710678118654752440f));
}

__global__ __launch_bounds__(256)
void gelu_layernorm(float* __restrict__ out,
                    const float* __restrict__ gamma,
                    const float* __restrict__ beta) {
    __shared__ float red[256];
    const int row = blockIdx.x;
    const int t   = threadIdx.x;
    float* rp = out + (size_t)row * HDIM;

    const float g0 = gelu_exact(rp[t]);
    const float g1 = gelu_exact(rp[t + 256]);

    red[t] = g0 + g1;
    __syncthreads();
#pragma unroll
    for (int s = 128; s >= 1; s >>= 1) {
        if (t < s) red[t] += red[t + s];
        __syncthreads();
    }
    const float mu = red[0] * (1.0f / (float)HDIM);
    __syncthreads();

    const float d0 = g0 - mu;
    const float d1 = g1 - mu;
    red[t] = d0 * d0 + d1 * d1;
    __syncthreads();
#pragma unroll
    for (int s = 128; s >= 1; s >>= 1) {
        if (t < s) red[t] += red[t + s];
        __syncthreads();
    }
    const float var = red[0] * (1.0f / (float)HDIM);
    const float rs  = __frsqrt_rn(var + LN_EPS);

    rp[t]       = d0 * rs * gamma[t]       + beta[t];
    rp[t + 256] = d1 * rs * gamma[t + 256] + beta[t + 256];
}

// -------------------------------------------------------------------------
// gene_idx = arange(20000), int32, appended after the [4096,512] output.
// -------------------------------------------------------------------------
__global__ void write_iota(int* __restrict__ gi, int n) {
    const int i = blockIdx.x * blockDim.x + threadIdx.x;
    if (i < n) gi[i] = i;
}

extern "C" void kernel_launch(void* const* d_in, const int* in_sizes, int n_in,
                              void* d_out, int out_size, void* d_ws, size_t ws_size,
                              hipStream_t stream) {
    (void)in_sizes; (void)n_in; (void)out_size; (void)d_ws; (void)ws_size;
    const float* x     = (const float*)d_in[0];
    const float* emb   = (const float*)d_in[1];
    const float* gamma = (const float*)d_in[2];
    const float* beta  = (const float*)d_in[3];
    float* out = (float*)d_out;

    gemm_xemb_bf16<<<dim3((BDIM / BM) * (HDIM / BN)), dim3(256), 0, stream>>>(x, emb, out);
    gelu_layernorm<<<dim3(BDIM), dim3(256), 0, stream>>>(out, gamma, beta);
    write_iota<<<dim3((GDIM + 255) / 256), dim3(256), 0, stream>>>(
        (int*)(out + (size_t)BDIM * HDIM), GDIM);
}
